// AutoCalibrationInjector_23742579212617
// MI455X (gfx1250) — matrix operand order repaired and checked by reading, not compiled, not run on hardware
//
#include <hip/hip_runtime.h>
#include <hip/hip_bf16.h>

// ---------------------------------------------------------------------------
// Problem dims (fixed by the reference)
// ---------------------------------------------------------------------------
enum : int { B_ = 4, S_ = 4096, E_ = 1024, H_ = 16, M_ = 512, D_ = 64 };
enum : int { BS_ = B_ * S_, HD_ = H_ * D_ };          // 16384, 1024
static const float kScale = 0.03125f;                 // 1/sqrt(E)
static const float kInvBS = 1.0f / (float)BS_;

typedef __bf16 bf16_t;
typedef bf16_t v16bf __attribute__((ext_vector_type(16)));
typedef float  v8f   __attribute__((ext_vector_type(8)));

union FragU { v16bf v; int4 q[2]; bf16_t e[16]; };

// B / A (bf16 source) fragment: two 16-byte chunks at element offsets off0/off1
__device__ __forceinline__ v16bf load_frag_b(const bf16_t* p, int off0, int off1) {
  FragU u;
  u.q[0] = *(const int4*)(p + off0);
  u.q[1] = *(const int4*)(p + off1);
  return u.v;
}

__device__ __forceinline__ v16bf cvt_frag(float4 a, float4 b, float4 c, float4 d) {
  FragU u;
  u.e[0]  = (bf16_t)a.x; u.e[1]  = (bf16_t)a.y; u.e[2]  = (bf16_t)a.z; u.e[3]  = (bf16_t)a.w;
  u.e[4]  = (bf16_t)b.x; u.e[5]  = (bf16_t)b.y; u.e[6]  = (bf16_t)b.z; u.e[7]  = (bf16_t)b.w;
  u.e[8]  = (bf16_t)c.x; u.e[9]  = (bf16_t)c.y; u.e[10] = (bf16_t)c.z; u.e[11] = (bf16_t)c.w;
  u.e[12] = (bf16_t)d.x; u.e[13] = (bf16_t)d.y; u.e[14] = (bf16_t)d.z; u.e[15] = (bf16_t)d.w;
  return u.v;
}

// A fragment from an f32 row (row-major, K contiguous).  ISA A layout (16-bit,
// 16x32): lanes 0-15 hold K = 0..7 & 16..23, lanes 16-31 hold K = 8..15 & 24..31.
// half = (lane>>4)*8.
__device__ __forceinline__ v16bf load_a_f32(const float* p, int half) {
  const float4* p0 = (const float4*)(p + half);
  const float4* p1 = (const float4*)(p + 16 + half);
  return cvt_frag(p0[0], p0[1], p1[0], p1[1]);
}

__device__ __forceinline__ v8f wmma_bf16(v16bf a, v16bf b, v8f c) {
  return __builtin_amdgcn_wmma_f32_16x16x32_bf16(false, a, false, b, (short)0, c,
                                                 false, false);
}

// ---------------------------------------------------------------------------
// Prep kernels: bf16 casts / transposes of the small parameter tensors
// ---------------------------------------------------------------------------
// out[n*K + k] = in[k*N + n]   (both K = N = 1024 here)
__global__ void k_transpose_cvt(const float* __restrict__ in, bf16_t* __restrict__ out,
                                int K, int N) {
  int idx = blockIdx.x * 256 + threadIdx.x;
  if (idx >= K * N) return;
  int n = idx / K, k = idx - n * K;
  out[idx] = (bf16_t)in[(size_t)k * N + n];
}

__global__ void k_kcal(const float* __restrict__ kp, const float* __restrict__ cal,
                       bf16_t* __restrict__ out, int n) {
  int idx = blockIdx.x * 256 + threadIdx.x;
  if (idx < n) out[idx] = (bf16_t)(kp[idx] + cal[idx]);
}

// value [H][M][D] -> valT [H][D][M]
__global__ void k_valT(const float* __restrict__ val, bf16_t* __restrict__ out) {
  int idx = blockIdx.x * 256 + threadIdx.x;   // over H*D*M = 524288
  if (idx >= H_ * D_ * M_) return;
  int h = idx >> 15, rem = idx & 32767;
  int d = rem >> 9, m = rem & 511;
  out[idx] = (bf16_t)val[(size_t)(h << 15) + m * D_ + d];
}

__global__ void k_zero(float* __restrict__ p, int n) {
  int idx = blockIdx.x * 256 + threadIdx.x;
  if (idx < n) p[idx] = 0.0f;
}

__global__ void k_finalize_act(const float* __restrict__ activity,
                               const float* __restrict__ acc,
                               float* __restrict__ out) {
  int idx = blockIdx.x * 256 + threadIdx.x;
  if (idx < H_ * M_)
    out[idx] = activity[idx] * 0.999f + 0.001f * (acc[idx] * kInvBS);
}

// ---------------------------------------------------------------------------
// Register-blocked WMMA GEMM:  C[M,N] = A[M,K] * Bt[N,K]^T + bias
// 256 threads / block = 8 waves in a 2x4 grid; each wave owns a 64x64 tile
// (4x4 WMMA tiles).  Fragments load straight from global (weights stay in L2).
// ---------------------------------------------------------------------------
template <bool AF32, bool CF32>
__global__ __launch_bounds__(256) void gemm_bf16_wmma(
    const void* __restrict__ Ap, const bf16_t* __restrict__ Bt,
    const float* __restrict__ bias, void* __restrict__ Cp,
    int M, int N, int K) {
  const int lane = threadIdx.x & 31;
  const int wid  = threadIdx.x >> 5;
  const int wm = wid & 1, wn = wid >> 1;
  const int lr = lane & 15, lh = lane >> 4;
  const int m0 = blockIdx.x * 128 + wm * 64;
  const int n0 = blockIdx.y * 256 + wn * 64;

  v8f acc[4][4];
  const v8f vz = {0.f, 0.f, 0.f, 0.f, 0.f, 0.f, 0.f, 0.f};
#pragma unroll
  for (int i = 0; i < 4; ++i)
#pragma unroll
    for (int j = 0; j < 4; ++j) acc[i][j] = vz;

  for (int k0 = 0; k0 < K; k0 += 32) {
    v16bf a[4], b[4];
#pragma unroll
    for (int i = 0; i < 4; ++i) {
      const size_t row = (size_t)(m0 + i * 16 + lr);
      if constexpr (AF32)
        a[i] = load_a_f32((const float*)Ap + row * K + k0, lh * 8);
      else
        a[i] = load_frag_b((const bf16_t*)Ap + row * K + k0, lh * 8, 16 + lh * 8);
    }
#pragma unroll
    for (int j = 0; j < 4; ++j)
      b[j] = load_frag_b(Bt + (size_t)(n0 + j * 16 + lr) * K + k0,
                         lh * 8, 16 + lh * 8);
#pragma unroll
    for (int i = 0; i < 4; ++i)
#pragma unroll
      for (int j = 0; j < 4; ++j) acc[i][j] = wmma_bf16(a[i], b[j], acc[i][j]);
  }

  // C/D layout: N = lane%16, M = e + 8*(lane>>4)
#pragma unroll
  for (int j = 0; j < 4; ++j) {
    const int col = n0 + j * 16 + lr;
    const float bz = bias[col];
#pragma unroll
    for (int i = 0; i < 4; ++i) {
#pragma unroll
      for (int e = 0; e < 8; ++e) {
        const size_t row = (size_t)(m0 + i * 16 + lh * 8 + e);
        const float v = acc[i][j][e] + bz;
        if constexpr (CF32)
          ((float*)Cp)[row * N + col] = v;
        else
          ((bf16_t*)Cp)[row * N + col] = (bf16_t)v;
      }
    }
  }
}

// ---------------------------------------------------------------------------
// Fused attention: one wave per (16-row tile, head).
//   scores (WMMA, f32 in LDS) -> softmax -> activity atomics -> ctx (WMMA)
// ---------------------------------------------------------------------------
__global__ __launch_bounds__(32) void attn_kernel(
    const bf16_t* __restrict__ q, const bf16_t* __restrict__ kcal,
    const bf16_t* __restrict__ valT, bf16_t* __restrict__ ctx,
    float* __restrict__ act_acc) {
  __shared__ float sc[16 * M_];   // 32 KB score / attn tile
  __shared__ float red[32];

  const int lane = threadIdx.x;
  const int lr = lane & 15, lh = lane >> 4;
  const int row0 = blockIdx.x * 16;        // flat (b,s) row base
  const int h = blockIdx.y;

  // ---- score = q . k^T * scale -------------------------------------------
  const bf16_t* qp = q + (size_t)(row0 + lr) * HD_ + h * D_;
  const v16bf aq0 = load_frag_b(qp,      lh * 8, 16 + lh * 8);   // K = 0..31
  const v16bf aq1 = load_frag_b(qp + 32, lh * 8, 16 + lh * 8);   // K = 32..63
  const bf16_t* kp = kcal + (size_t)h * M_ * D_;

  for (int mc = 0; mc < M_ / 16; ++mc) {
    const bf16_t* bp = kp + (size_t)(mc * 16 + lr) * D_;
    v16bf b0 = load_frag_b(bp,      lh * 8, 16 + lh * 8);
    v16bf b1 = load_frag_b(bp + 32, lh * 8, 16 + lh * 8);
    v8f s = {0.f, 0.f, 0.f, 0.f, 0.f, 0.f, 0.f, 0.f};
    s = wmma_bf16(aq0, b0, s);
    s = wmma_bf16(aq1, b1, s);
#pragma unroll
    for (int e = 0; e < 8; ++e)
      sc[(e + lh * 8) * M_ + mc * 16 + lr] = s[e] * kScale;
  }
  __syncthreads();

  // ---- softmax over M=512 (2 lanes per row, half row each) ---------------
  float* rowp = &sc[lr * M_ + lh * 256];
  float pm = -3.4e38f;
  for (int j = 0; j < 256; ++j) pm = fmaxf(pm, rowp[j]);
  red[lane] = pm;
  __syncthreads();
  const float rm = fmaxf(red[lr], red[lr + 16]);
  __syncthreads();
  float ps = 0.f;
  for (int j = 0; j < 256; ++j) {
    const float ev = __expf(rowp[j] - rm);
    rowp[j] = ev;
    ps += ev;
  }
  red[lane] = ps;
  __syncthreads();
  const float inv = 1.0f / (red[lr] + red[lr + 16]);
  for (int j = 0; j < 256; ++j) rowp[j] *= inv;
  __syncthreads();

  // ---- activity accumulation (sum over the 16 rows of this tile) ---------
  for (int m = lane; m < M_; m += 32) {
    float s = 0.f;
#pragma unroll
    for (int r = 0; r < 16; ++r) s += sc[r * M_ + m];
    atomicAdd(&act_acc[h * M_ + m], s);
  }

  // ---- ctx = attn . value  (A from LDS, B = valT[h][D][M]) ---------------
  const bf16_t* vp = valT + (size_t)h * D_ * M_;
  v8f c[4];
  const v8f vz = {0.f, 0.f, 0.f, 0.f, 0.f, 0.f, 0.f, 0.f};
#pragma unroll
  for (int t = 0; t < 4; ++t) c[t] = vz;

  for (int ks = 0; ks < M_ / 32; ++ks) {
    const int k0 = ks * 32;
    const v16bf a = load_a_f32(&sc[lr * M_ + k0], lh * 8);   // ds_load_b128 + cvt
#pragma unroll
    for (int t = 0; t < 4; ++t) {
      const v16bf b = load_frag_b(vp + (size_t)(t * 16 + lr) * M_ + k0,
                                  lh * 8, 16 + lh * 8);
      c[t] = wmma_bf16(a, b, c[t]);
    }
  }
#pragma unroll
  for (int t = 0; t < 4; ++t)
#pragma unroll
    for (int e = 0; e < 8; ++e)
      ctx[(size_t)(row0 + lh * 8 + e) * HD_ + h * D_ + t * 16 + lr] =
          (bf16_t)c[t][e];
}

// ---------------------------------------------------------------------------
// Host launcher
// ---------------------------------------------------------------------------
extern "C" void kernel_launch(void* const* d_in, const int* in_sizes, int n_in,
                              void* d_out, int out_size, void* d_ws, size_t ws_size,
                              hipStream_t stream) {
  const float* x        = (const float*)d_in[0];
  const float* Wq       = (const float*)d_in[1];
  const float* bq       = (const float*)d_in[2];
  const float* keyp     = (const float*)d_in[3];
  const float* cal      = (const float*)d_in[4];
  const float* val      = (const float*)d_in[5];
  const float* Wd       = (const float*)d_in[6];
  const float* bd       = (const float*)d_in[7];
  const float* activity = (const float*)d_in[8];
  float* out = (float*)d_out;

  char* ws = (char*)d_ws;
  bf16_t* WqT  = (bf16_t*)(ws);                         //  2 MB [HD][E]
  bf16_t* WdT  = (bf16_t*)(ws + (size_t)2097152);       //  2 MB [E][HD]
  bf16_t* kcal = (bf16_t*)(ws + (size_t)4194304);       //  1 MB [H][M][D]
  bf16_t* valT = (bf16_t*)(ws + (size_t)5242880);       //  1 MB [H][D][M]
  bf16_t* qbf  = (bf16_t*)(ws + (size_t)6291456);       // 32 MB [BS][HD]
  bf16_t* ctxb = (bf16_t*)(ws + (size_t)39845888);      // 32 MB [BS][HD]
  float*  acc  = (float* )(ws + (size_t)73400320);      // 32 KB [H][M]

  // --- prep: bf16 casts / transposes + zero activity accumulator ----------
  k_transpose_cvt<<<(E_ * HD_ + 255) / 256, 256, 0, stream>>>(Wq, WqT, E_, HD_);
  k_transpose_cvt<<<(E_ * HD_ + 255) / 256, 256, 0, stream>>>(Wd, WdT, HD_, E_);
  k_kcal<<<(H_ * M_ * D_ + 255) / 256, 256, 0, stream>>>(keyp, cal, kcal,
                                                          H_ * M_ * D_);
  k_valT<<<(H_ * D_ * M_ + 255) / 256, 256, 0, stream>>>(val, valT);
  k_zero<<<(H_ * M_ + 255) / 256, 256, 0, stream>>>(acc, H_ * M_);

  // --- q = x * Wq + bq  (A = f32 x, C = bf16 q) ---------------------------
  gemm_bf16_wmma<true, false>
      <<<dim3(BS_ / 128, HD_ / 256), 256, 0, stream>>>(x, WqT, bq, qbf,
                                                       BS_, HD_, E_);

  // --- fused score/softmax/activity/ctx -----------------------------------
  attn_kernel<<<dim3(BS_ / 16, H_), 32, 0, stream>>>(qbf, kcal, valT, ctxb, acc);

  // --- out = ctx * Wd + bd  (A = bf16 ctx, C = f32 out) -------------------
  gemm_bf16_wmma<false, true>
      <<<dim3(BS_ / 128, E_ / 256), 256, 0, stream>>>(ctxb, WdT, bd, out,
                                                      BS_, E_, HD_);

  // --- new_activity appended after out ------------------------------------
  k_finalize_act<<<(H_ * M_ + 255) / 256, 256, 0, stream>>>(
      activity, acc, out + (size_t)BS_ * E_);
}
